// GAT_VGAE_2869038153804
// MI455X (gfx1250) — matrix operand order repaired and checked by reading, not compiled, not run on hardware
//
#include <hip/hip_runtime.h>
#include <cstddef>

// ---------------- problem constants ----------------
constexpr int   kN    = 2048;
constexpr int   kFIN  = 256;
constexpr int   kC1   = 128;
constexpr int   kH    = 4;
constexpr int   kHC1  = kH * kC1;       // 512
constexpr int   kEMB  = 64;
constexpr int   kE    = 65536;
constexpr int   kET   = kE + kN;        // edges + self loops = 67584
constexpr long long kNN = (long long)kN * (long long)kN; // 4194304
constexpr float kSlope = 0.2f;

typedef __attribute__((ext_vector_type(16))) _Float16 v16h;
typedef __attribute__((ext_vector_type(8)))  _Float16 v8h;
typedef __attribute__((ext_vector_type(8)))  float    v8f;
typedef __attribute__((ext_vector_type(4)))  float    v4f;

// ---------------- helpers ----------------
__global__ void k_f32_to_f16(const float* __restrict__ in, _Float16* __restrict__ out, int n) {
  int i = blockIdx.x * blockDim.x + threadIdx.x;
  if (i < n) out[i] = (_Float16)in[i];
}

// in[rows,cols] (row-major f32) -> out[cols,rows] (row-major f16)  (weight transpose)
__global__ void k_f32_to_f16_T(const float* __restrict__ in, _Float16* __restrict__ out,
                               int rows, int cols) {
  int i = blockIdx.x * blockDim.x + threadIdx.x;
  if (i < rows * cols) {
    int r = i / cols, c = i - r * cols;
    out[(size_t)c * rows + r] = (_Float16)in[i];
  }
}

__global__ void k_fill(float* __restrict__ p, float v, int n) {
  int i = blockIdx.x * blockDim.x + threadIdx.x;
  if (i < n) p[i] = v;
}

__device__ __forceinline__ void edge_nodes(const int* __restrict__ ei, int e, int& src, int& dst) {
  if (e < kE) { src = ei[e]; dst = ei[kE + e]; }
  else        { src = dst = e - kE; }          // appended self loops
}

__device__ __forceinline__ void atomicMaxF(float* addr, float val) {
  int* ai = (int*)addr;
  int old = __float_as_int(*addr);
  while (__int_as_float(old) < val) {
    int prev = atomicCAS(ai, old, __float_as_int(val));
    if (prev == old) break;
    old = prev;
  }
}

// ---------------- WMMA GEMM: C[M,N] = A[M,K](f16) * BT[N,K](f16), f32 accum ----------------
// One wave (32 lanes) per 16x16 C tile; K fully unrolled in steps of 32 via
// v_wmma_f32_16x16x32_f16. B is pre-transposed so both fragments load with b128s.
template <int KDIM>
__global__ void k_gemm_wmma_f16(const _Float16* __restrict__ A,
                                const _Float16* __restrict__ BT,
                                float* __restrict__ C,
                                int Ndim) {
  const int lane = threadIdx.x;        // 0..31, wave32
  const int m    = lane & 15;
  const int hi   = lane >> 4;          // lane half selects contiguous K half (k = 16*hi + s)
  const int tn   = blockIdx.x * 16;
  const int tm   = blockIdx.y * 16;

  const _Float16* Arow = A  + (size_t)(tm + m) * KDIM;
  const _Float16* Brow = BT + (size_t)(tn + m) * KDIM;

  v8f acc = {};
#pragma unroll
  for (int k0 = 0; k0 < KDIM; k0 += 32) {
    // 16-bit A 16x32 fragment: lanes 0-15 -> K=k0..k0+15 ; lanes 16-31 -> +16
    const int ka = k0 + hi * 16;
    v16h a = *(const v16h*)(Arow + ka);
    // 16-bit B 32x16 fragment: lanes 0-15 hold K=k0..k0+15, lanes 16-31 -> +16.
    // With BT row-major, that is 16 contiguous f16 -> one 32B vector load.
    const int kb = k0 + hi * 16;
    v16h b = *(const v16h*)(Brow + kb);

    acc = __builtin_amdgcn_wmma_f32_16x16x32_f16(false, a, false, b,
                                                 (short)0, acc, false, false);
  }
  // C/D layout: VGPR r -> row (hi*8 + r), col = lane&15
#pragma unroll
  for (int r = 0; r < 8; ++r)
    C[(size_t)(tm + hi * 8 + r) * (size_t)Ndim + (size_t)(tn + m)] = acc[r];
}

// ---------------- GAT attention pieces ----------------
__global__ void k_att1(const float* __restrict__ h1, const float* __restrict__ as_w,
                       const float* __restrict__ ad_w, float* __restrict__ a_src,
                       float* __restrict__ a_dst) {
  int t = blockIdx.x * blockDim.x + threadIdx.x;  // n*H + h
  if (t >= kN * kH) return;
  int n = t >> 2, h = t & 3;
  const float* row = h1 + (size_t)n * kHC1 + h * kC1;
  const float* ws  = as_w + h * kC1;
  const float* wd  = ad_w + h * kC1;
  float s = 0.f, d = 0.f;
  for (int c = 0; c < kC1; ++c) { float v = row[c]; s += v * ws[c]; d += v * wd[c]; }
  a_src[t] = s; a_dst[t] = d;
}

__global__ void k_att2(const float* __restrict__ h2, const float* __restrict__ as_w,
                       const float* __restrict__ ad_w, float* __restrict__ a_src,
                       float* __restrict__ a_dst) {
  int n = blockIdx.x * blockDim.x + threadIdx.x;
  if (n >= kN) return;
  const float* row = h2 + (size_t)n * kEMB;
  float s = 0.f, d = 0.f;
  for (int c = 0; c < kEMB; ++c) { float v = row[c]; s += v * as_w[c]; d += v * ad_w[c]; }
  a_src[n] = s; a_dst[n] = d;
}

__global__ void k_edge_max(const int* __restrict__ ei, const float* __restrict__ a_src,
                           const float* __restrict__ a_dst, float* __restrict__ mx, int heads) {
  int t = blockIdx.x * blockDim.x + threadIdx.x;
  if (t >= kET * heads) return;
  int e = t / heads, h = t - e * heads;
  int src, dst; edge_nodes(ei, e, src, dst);
  float l = a_src[src * heads + h] + a_dst[dst * heads + h];
  l = (l > 0.f) ? l : kSlope * l;
  atomicMaxF(&mx[dst * heads + h], l);
}

__global__ void k_edge_expsum(const int* __restrict__ ei, const float* __restrict__ a_src,
                              const float* __restrict__ a_dst, const float* __restrict__ mx,
                              float* __restrict__ alpha, float* __restrict__ ssum, int heads) {
  int t = blockIdx.x * blockDim.x + threadIdx.x;
  if (t >= kET * heads) return;
  int e = t / heads, h = t - e * heads;
  int src, dst; edge_nodes(ei, e, src, dst);
  float l = a_src[src * heads + h] + a_dst[dst * heads + h];
  l = (l > 0.f) ? l : kSlope * l;
  float w = expf(l - mx[dst * heads + h]);
  alpha[t] = w;
  atomicAdd(&ssum[dst * heads + h], w);
}

// ---------------- message aggregation (scatter-add) ----------------
__global__ void k_agg1(const int* __restrict__ ei, const float* __restrict__ h1,
                       const float* __restrict__ alpha, const float* __restrict__ ssum,
                       float* __restrict__ acc) {
  int e = blockIdx.x;
  int src, dst; edge_nodes(ei, e, src, dst);
  for (int i = threadIdx.x; i < kHC1; i += blockDim.x) {
    int h = i >> 7;                                   // kC1 == 128
    float coef = alpha[e * kH + h] / (ssum[dst * kH + h] + 1e-16f);
    atomicAdd(&acc[(size_t)dst * kHC1 + i], h1[(size_t)src * kHC1 + i] * coef);
  }
}

__global__ void k_agg2(const int* __restrict__ ei, const float* __restrict__ h2,
                       const float* __restrict__ alpha, const float* __restrict__ ssum,
                       float* __restrict__ acc) {
  int e = blockIdx.x;
  int c = threadIdx.x;                                 // blockDim == kEMB
  int src, dst; edge_nodes(ei, e, src, dst);
  float coef = alpha[e] / (ssum[dst] + 1e-16f);
  atomicAdd(&acc[(size_t)dst * kEMB + c], h2[(size_t)src * kEMB + c] * coef);
}

__global__ void k_bias_relu_h(const float* __restrict__ acc, const float* __restrict__ b,
                              _Float16* __restrict__ out, int n) {
  int i = blockIdx.x * blockDim.x + threadIdx.x;
  if (i < n) {
    float v = acc[i] + b[i & (kHC1 - 1)];
    out[i] = (_Float16)(v > 0.f ? v : 0.f);
  }
}

// ---------------- reparameterize + mean over nodes ----------------
__global__ void k_node_z(const float* __restrict__ emb_acc, const float* __restrict__ b2,
                         const float* __restrict__ Wmu, const float* __restrict__ bmu,
                         const float* __restrict__ Wlv, const float* __restrict__ blv,
                         const float* __restrict__ eps, float* __restrict__ zm) {
  __shared__ float er[kEMB];
  int n = blockIdx.x, c = threadIdx.x;                 // blockDim == kEMB
  er[c] = emb_acc[(size_t)n * kEMB + c] + b2[c];
  __syncthreads();
  float mu = bmu[c], lv = blv[c];
  for (int k = 0; k < kEMB; ++k) {
    float v = er[k];
    mu += v * Wmu[k * kEMB + c];
    lv += v * Wlv[k * kEMB + c];
  }
  float z = mu + eps[(size_t)n * kEMB + c] * expf(0.5f * lv);
  atomicAdd(&zm[c], z * (1.0f / (float)kN));
}

// ---------------- streaming decoder: sigmoid(zm @ Wd + bd) ----------------
// 1.07 GB single-use read of Wd -> non-temporal loads, float4 vectorized.
__global__ void k_decoder(const float* __restrict__ zm, const float* __restrict__ Wd,
                          const float* __restrict__ bd, float* __restrict__ out) {
  __shared__ float zs[kEMB];
  if (threadIdx.x < kEMB) zs[threadIdx.x] = zm[threadIdx.x];
  __syncthreads();
  size_t j = ((size_t)blockIdx.x * blockDim.x + threadIdx.x) * 4;
  v4f acc = *(const v4f*)(bd + j);
#pragma unroll 8
  for (int k = 0; k < kEMB; ++k) {
    v4f w = __builtin_nontemporal_load((const v4f*)(Wd + (size_t)k * (size_t)kNN + j));
    acc += zs[k] * w;
  }
  v4f r;
#pragma unroll
  for (int q = 0; q < 4; ++q) r[q] = 1.0f / (1.0f + expf(-acc[q]));
  __builtin_nontemporal_store(r, (v4f*)(out + j));
}

// ---------------- host orchestration ----------------
extern "C" void kernel_launch(void* const* d_in, const int* in_sizes, int n_in,
                              void* d_out, int out_size, void* d_ws, size_t ws_size,
                              hipStream_t stream) {
  (void)in_sizes; (void)n_in; (void)out_size; (void)ws_size;
  const int*   ei  = (const int*)  d_in[0];
  const float* x   = (const float*)d_in[1];
  const float* eps = (const float*)d_in[2];
  const float* W1  = (const float*)d_in[3];
  const float* as1 = (const float*)d_in[4];
  const float* ad1 = (const float*)d_in[5];
  const float* b1  = (const float*)d_in[6];
  const float* W2  = (const float*)d_in[7];
  const float* as2 = (const float*)d_in[8];
  const float* ad2 = (const float*)d_in[9];
  const float* b2  = (const float*)d_in[10];
  const float* Wmu = (const float*)d_in[11];
  const float* bmu = (const float*)d_in[12];
  const float* Wlv = (const float*)d_in[13];
  const float* blv = (const float*)d_in[14];
  const float* Wd  = (const float*)d_in[15];
  const float* bd  = (const float*)d_in[16];
  float* out = (float*)d_out;

  char* ws = (char*)d_ws;
  size_t off = 0;
  auto alloc = [&](size_t bytes) -> void* {
    void* p = ws + off; off += (bytes + 255) & ~(size_t)255; return p;
  };

  _Float16* x_h   = (_Float16*)alloc((size_t)kN * kFIN * 2);
  _Float16* W1t_h = (_Float16*)alloc((size_t)kFIN * kHC1 * 2);   // [512,256] = W1^T
  _Float16* W2t_h = (_Float16*)alloc((size_t)kHC1 * kEMB * 2);   // [64,512]  = W2^T
  float* h1       = (float*)alloc((size_t)kN * kHC1 * 4);
  float* asrc1    = (float*)alloc((size_t)kN * kH * 4);
  float* adst1    = (float*)alloc((size_t)kN * kH * 4);
  float* m1       = (float*)alloc((size_t)kN * kH * 4);
  float* s1       = (float*)alloc((size_t)kN * kH * 4);
  float* alpha1   = (float*)alloc((size_t)kET * kH * 4);
  float* hacc     = (float*)alloc((size_t)kN * kHC1 * 4);
  _Float16* hid_h = (_Float16*)alloc((size_t)kN * kHC1 * 2);
  float* h2       = (float*)alloc((size_t)kN * kEMB * 4);
  float* asrc2    = (float*)alloc((size_t)kN * 4);
  float* adst2    = (float*)alloc((size_t)kN * 4);
  float* m2       = (float*)alloc((size_t)kN * 4);
  float* s2       = (float*)alloc((size_t)kN * 4);
  float* alpha2   = (float*)alloc((size_t)kET * 4);
  float* eacc     = (float*)alloc((size_t)kN * kEMB * 4);
  float* zm       = (float*)alloc(256);

  auto cdiv = [](int a, int b) { return (a + b - 1) / b; };
  const float NEG = -3.402823466e38f;

  // convert + transpose weights + init accumulators
  k_f32_to_f16<<<cdiv(kN * kFIN, 256), 256, 0, stream>>>(x, x_h, kN * kFIN);
  k_f32_to_f16_T<<<cdiv(kFIN * kHC1, 256), 256, 0, stream>>>(W1, W1t_h, kFIN, kHC1);
  k_f32_to_f16_T<<<cdiv(kHC1 * kEMB, 256), 256, 0, stream>>>(W2, W2t_h, kHC1, kEMB);
  k_fill<<<cdiv(kN * kHC1, 256), 256, 0, stream>>>(hacc, 0.f, kN * kHC1);
  k_fill<<<cdiv(kN * kEMB, 256), 256, 0, stream>>>(eacc, 0.f, kN * kEMB);
  k_fill<<<cdiv(kN * kH, 256), 256, 0, stream>>>(s1, 0.f, kN * kH);
  k_fill<<<cdiv(kN, 256), 256, 0, stream>>>(s2, 0.f, kN);
  k_fill<<<1, 64, 0, stream>>>(zm, 0.f, kEMB);
  k_fill<<<cdiv(kN * kH, 256), 256, 0, stream>>>(m1, NEG, kN * kH);
  k_fill<<<cdiv(kN, 256), 256, 0, stream>>>(m2, NEG, kN);

  // ---- GAT layer 1 (H=4, C=128, concat) ----
  k_gemm_wmma_f16<kFIN><<<dim3(kHC1 / 16, kN / 16), 32, 0, stream>>>(x_h, W1t_h, h1, kHC1);
  k_att1<<<cdiv(kN * kH, 256), 256, 0, stream>>>(h1, as1, ad1, asrc1, adst1);
  k_edge_max<<<cdiv(kET * kH, 256), 256, 0, stream>>>(ei, asrc1, adst1, m1, kH);
  k_edge_expsum<<<cdiv(kET * kH, 256), 256, 0, stream>>>(ei, asrc1, adst1, m1, alpha1, s1, kH);
  k_agg1<<<kET, 256, 0, stream>>>(ei, h1, alpha1, s1, hacc);
  k_bias_relu_h<<<cdiv(kN * kHC1, 256), 256, 0, stream>>>(hacc, b1, hid_h, kN * kHC1);

  // ---- GAT layer 2 (H=1, C=64, mean==identity) ----
  k_gemm_wmma_f16<kHC1><<<dim3(kEMB / 16, kN / 16), 32, 0, stream>>>(hid_h, W2t_h, h2, kEMB);
  k_att2<<<cdiv(kN, 256), 256, 0, stream>>>(h2, as2, ad2, asrc2, adst2);
  k_edge_max<<<cdiv(kET, 256), 256, 0, stream>>>(ei, asrc2, adst2, m2, 1);
  k_edge_expsum<<<cdiv(kET, 256), 256, 0, stream>>>(ei, asrc2, adst2, m2, alpha2, s2, 1);
  k_agg2<<<kET, kEMB, 0, stream>>>(ei, h2, alpha2, s2, eacc);

  // ---- VAE head: mu/logvar, reparameterize, mean over nodes ----
  k_node_z<<<kN, kEMB, 0, stream>>>(eacc, b2, Wmu, bmu, Wlv, blv, eps, zm);

  // ---- streaming decoder (bandwidth-dominant: ~1.07 GB NT read) ----
  k_decoder<<<(unsigned)(kNN / (256 * 4)), 256, 0, stream>>>(zm, Wd, bd, out);
}
